// MultiheadAttention_80753975099589
// MI455X (gfx1250) — matrix operand order repaired and checked by reading, not compiled, not run on hardware
//
#include <hip/hip_runtime.h>

#define D_MODEL 768
#define N_HEADS 12
#define HEAD_DIM 64
#define BATCH 2
#define SEQ 2048
#define ROWS (BATCH * SEQ) /* 4096 */
#define ROWL 2052          /* LDS row stride (floats): 2052 % 64 = 4 -> conflict-free halves */

typedef __attribute__((ext_vector_type(16))) __bf16 bf16x16;
typedef __attribute__((ext_vector_type(8)))  float  floatx8;

// ---------- tile loaders ----------
// A-matrix tile (16 x 32 bf16) from row-major bf16 [row][k].
// Lane layout (ISA 7.12.2): lane&15 = M, lane>>4 = K-half; elements 0..7 -> k0+8*half+e,
// 8..15 -> k0+16+8*half+(e-8): two contiguous 16-byte runs -> two b128 loads.
__device__ __forceinline__ bf16x16 load_bf_A(const __bf16* __restrict__ p, int r0, int ld,
                                             int k0, int lane) {
    const int kh = lane >> 4;
    const __bf16* base = p + (size_t)(r0 + (lane & 15)) * (size_t)ld + k0 + kh * 8;
    union { bf16x16 v; uint4 q[2]; } t;
    t.q[0] = *reinterpret_cast<const uint4*>(base);
    t.q[1] = *reinterpret_cast<const uint4*>(base + 16);
    return t.v;
}

// B-matrix tile (32 x 16 bf16) with B[k][n] = W[n][k], W row-major bf16 [n][k].
// Lane layout: lane&15 = N, lane>>4 = K-half; elements 0..7 -> k0+8*half+e, 8..15 -> k0+16+8*half+(e-8) -> two b128 loads.
__device__ __forceinline__ bf16x16 load_bf_B(const __bf16* __restrict__ p, int n0, int ld,
                                             int k0, int lane) {
    const int kh = lane >> 4;
    const __bf16* base = p + (size_t)(n0 + (lane & 15)) * (size_t)ld + k0 + kh * 8;
    union { bf16x16 v; uint4 q[2]; } t;
    t.q[0] = *reinterpret_cast<const uint4*>(base);
    t.q[1] = *reinterpret_cast<const uint4*>(base + 16);
    return t.v;
}

// A-matrix tile from f32 LDS scores (probs), native casts -> v_cvt_pk_bf16_f32.
__device__ __forceinline__ bf16x16 load_sc_A(const float* sc, int k0, int lane) {
    const int kh = lane >> 4;
    const float* base = sc + (lane & 15) * ROWL + k0 + kh * 8;
    bf16x16 v;
#pragma unroll
    for (int i = 0; i < 8; ++i) {
        v[i]     = (__bf16)base[i];
        v[i + 8] = (__bf16)base[16 + i];
    }
    return v;
}

__device__ __forceinline__ floatx8 wmma_bf16(bf16x16 a, bf16x16 b, floatx8 c) {
    return __builtin_amdgcn_wmma_f32_16x16x32_bf16(false, a, false, b, (short)0, c,
                                                   false, false);
}

// ---------- f32 -> bf16 bulk convert ----------
__global__ __launch_bounds__(256)
void cvt_f32_to_bf16(const float* __restrict__ src, __bf16* __restrict__ dst, int n8) {
    const int i = blockIdx.x * 256 + threadIdx.x;
    if (i >= n8) return;
    const float4* s = reinterpret_cast<const float4*>(src) + (size_t)i * 2;
    const float4 a = s[0], b = s[1];
    union { uint4 q; __bf16 h[8]; } t;
    t.h[0] = (__bf16)a.x; t.h[1] = (__bf16)a.y; t.h[2] = (__bf16)a.z; t.h[3] = (__bf16)a.w;
    t.h[4] = (__bf16)b.x; t.h[5] = (__bf16)b.y; t.h[6] = (__bf16)b.z; t.h[7] = (__bf16)b.w;
    reinterpret_cast<uint4*>(dst)[i] = t.q;
}

// ---------- generic projection: Y = A(MxK) @ W(NxK)^T + bias ----------
// mode 0: f32 row-major; mode 1: bf16 row-major; mode 2: bf16 transposed Y[n*ldy+m].
__global__ __launch_bounds__(256)
void gemm_xwT_bf16(const __bf16* __restrict__ A, int lda,
                   const __bf16* __restrict__ W, int ldw,
                   const float* __restrict__ bias,
                   void* __restrict__ Y, int ldy, int K, int mode) {
    const int lane = threadIdx.x & 31;
    const int wave = threadIdx.x >> 5;
    const int m0  = (blockIdx.x * 8 + wave) * 32;
    const int n0b = blockIdx.y * 64;

    floatx8 acc[2][4] = {};
    for (int k0 = 0; k0 < K; k0 += 32) {
        bf16x16 a0 = load_bf_A(A, m0,      lda, k0, lane);
        bf16x16 a1 = load_bf_A(A, m0 + 16, lda, k0, lane);
#pragma unroll
        for (int t = 0; t < 4; ++t) {
            bf16x16 b = load_bf_B(W, n0b + t * 16, ldw, k0, lane);
            acc[0][t] = wmma_bf16(a0, b, acc[0][t]);
            acc[1][t] = wmma_bf16(a1, b, acc[1][t]);
        }
    }
    const int kh = lane >> 4;
    const int nl = lane & 15;
#pragma unroll
    for (int mt = 0; mt < 2; ++mt) {
#pragma unroll
        for (int t = 0; t < 4; ++t) {
            const int n = n0b + t * 16 + nl;
            const float bb = bias ? bias[n] : 0.0f;
#pragma unroll
            for (int r = 0; r < 8; ++r) {
                const int m = m0 + mt * 16 + kh * 8 + r;
                const float v = acc[mt][t][r] + bb;
                if (mode == 0)      ((float*)Y)[(size_t)m * (size_t)ldy + n]  = v;
                else if (mode == 1) ((__bf16*)Y)[(size_t)m * (size_t)ldy + n] = (__bf16)v;
                else                ((__bf16*)Y)[(size_t)n * (size_t)ldy + m] = (__bf16)v;
            }
        }
    }
}

// ---------- fused attention: scores + softmax + PV for one (b, h, 16-query tile) ----------
// 8 waves; wave w owns keys [w*256, w*256+256). Scores live in LDS (16 x 2048 f32, padded),
// probs written to global exactly once, PV accumulated from LDS with wave K-split + LDS reduce.
__global__ __launch_bounds__(256)
void attn_fused(const __bf16* __restrict__ Q, const __bf16* __restrict__ Kc,
                const __bf16* __restrict__ Vt, const int* __restrict__ mask,
                float* __restrict__ probs, __bf16* __restrict__ AO) {
    __shared__ float sc[16 * ROWL];        // scores -> probs (f32)
    __shared__ float part[8 * 16 * 64];    // PV wave partials
    __shared__ float rowred[16 * 16];
    __shared__ float rowmax[16];
    __shared__ float rowsum[16];

    const int lane = threadIdx.x & 31;
    const int wave = threadIdx.x >> 5;
    const int kh = lane >> 4;
    const int nl = lane & 15;
    const int bh = blockIdx.z;
    const int bI = bh / N_HEADS;
    const int h  = bh % N_HEADS;
    const int q0 = blockIdx.x * 16;

    const __bf16* Qp = Q  + (size_t)bI * SEQ * D_MODEL + h * HEAD_DIM;
    const __bf16* Kp = Kc + (size_t)bI * SEQ * D_MODEL + h * HEAD_DIM;
    const __bf16* Vb = Vt + (size_t)h * HEAD_DIM * ROWS + (size_t)bI * SEQ;

    // ---- phase 1: scores = scale * Q K^T (masked) -> LDS ----
    const bf16x16 aq0 = load_bf_A(Qp, q0, D_MODEL, 0,  lane);
    const bf16x16 aq1 = load_bf_A(Qp, q0, D_MODEL, 32, lane);
#pragma unroll 4
    for (int t = 0; t < 16; ++t) {
        const int n0 = wave * 256 + t * 16;
        bf16x16 b0 = load_bf_B(Kp, n0, D_MODEL, 0,  lane);
        bf16x16 b1 = load_bf_B(Kp, n0, D_MODEL, 32, lane);
        floatx8 acc = {};
        acc = wmma_bf16(aq0, b0, acc);
        acc = wmma_bf16(aq1, b1, acc);
        const int kk = n0 + nl;
        const bool keep = mask[bI * SEQ + kk] != 0;
#pragma unroll
        for (int r = 0; r < 8; ++r)
            sc[(kh * 8 + r) * ROWL + kk] = keep ? acc[r] * 0.125f : -1e9f;
    }
    __syncthreads();

    // ---- phase 2: softmax over LDS rows; write probs to global once ----
    const int row = threadIdx.x >> 4;   // 0..15
    const int cl  = threadIdx.x & 15;   // 0..15
    float* srow = sc + row * ROWL;

    float mx = -3.4e38f;
    for (int c = cl; c < SEQ; c += 16) mx = fmaxf(mx, srow[c]);
    rowred[row * 16 + cl] = mx;
    __syncthreads();
    if (cl == 0) {
        float m2 = rowred[row * 16];
#pragma unroll
        for (int j = 1; j < 16; ++j) m2 = fmaxf(m2, rowred[row * 16 + j]);
        rowmax[row] = m2;
    }
    __syncthreads();
    mx = rowmax[row];

    float sum = 0.0f;
    for (int c = cl; c < SEQ; c += 16) {
        const float e = __expf(srow[c] - mx);
        srow[c] = e;
        sum += e;
    }
    rowred[row * 16 + cl] = sum;
    __syncthreads();
    if (cl == 0) {
        float s2 = 0.0f;
#pragma unroll
        for (int j = 0; j < 16; ++j) s2 += rowred[row * 16 + j];
        rowsum[row] = s2;
    }
    __syncthreads();
    const float inv = 1.0f / rowsum[row];
    float* Pg = probs + (size_t)bh * SEQ * SEQ + (size_t)(q0 + row) * SEQ;
    for (int c = cl; c < SEQ; c += 16) {
        const float pv = srow[c] * inv;
        srow[c] = pv;
        Pg[c] = pv;
    }
    __syncthreads();

    // ---- phase 3: PV with wave K-split (keys), A from LDS probs, B = V^T (bf16) ----
    floatx8 acc[4] = {};
#pragma unroll 2
    for (int ks = 0; ks < 8; ++ks) {
        const int kk = wave * 256 + ks * 32;
        bf16x16 a = load_sc_A(sc, kk, lane);
#pragma unroll
        for (int t = 0; t < 4; ++t) {
            bf16x16 b = load_bf_B(Vb, t * 16, ROWS, kk, lane);
            acc[t] = wmma_bf16(a, b, acc[t]);
        }
    }
#pragma unroll
    for (int t = 0; t < 4; ++t)
#pragma unroll
        for (int r = 0; r < 8; ++r)
            part[wave * 1024 + (kh * 8 + r) * 64 + t * 16 + nl] = acc[t][r];
    __syncthreads();

    for (int idx = threadIdx.x; idx < 1024; idx += 256) {
        float s = 0.0f;
#pragma unroll
        for (int w = 0; w < 8; ++w) s += part[w * 1024 + idx];
        const int m = idx >> 6;
        const int d = idx & 63;
        AO[((size_t)bI * SEQ + q0 + m) * D_MODEL + h * HEAD_DIM + d] = (__bf16)s;
    }
}

extern "C" void kernel_launch(void* const* d_in, const int* in_sizes, int n_in,
                              void* d_out, int out_size, void* d_ws, size_t ws_size,
                              hipStream_t stream) {
    const float* query = (const float*)d_in[0];
    const float* key_  = (const float*)d_in[1];
    const float* value = (const float*)d_in[2];
    const int*   mask  = (const int*)d_in[3];
    const float* wq = (const float*)d_in[4];
    const float* bq = (const float*)d_in[5];
    const float* wk = (const float*)d_in[6];
    const float* bk = (const float*)d_in[7];
    const float* wv = (const float*)d_in[8];
    const float* bv = (const float*)d_in[9];
    const float* wo = (const float*)d_in[10];
    const float* bo = (const float*)d_in[11];

    float* out   = (float*)d_out;                    // [ROWS, D_MODEL] f32
    float* probs = out + (size_t)ROWS * D_MODEL;     // [B*H, S, S] f32

    const size_t IN  = (size_t)ROWS * D_MODEL;
    const size_t WEL = (size_t)D_MODEL * D_MODEL;

    __bf16* xqb = (__bf16*)d_ws;
    __bf16* xkb = xqb + IN;
    __bf16* xvb = xkb + IN;
    __bf16* wqb = xvb + IN;
    __bf16* wkb = wqb + WEL;
    __bf16* wvb = wkb + WEL;
    __bf16* wob = wvb + WEL;
    __bf16* qb  = wob + WEL;     // [ROWS][D_MODEL]
    __bf16* kb  = qb  + IN;      // [ROWS][D_MODEL]
    __bf16* vtb = kb  + IN;      // [D_MODEL][ROWS] (transposed)
    __bf16* aob = vtb + IN;      // [ROWS][D_MODEL]

    const dim3 blk(256);

    const int in8 = (int)(IN / 8), w8 = (int)(WEL / 8);
    const dim3 gin((in8 + 255) / 256), gw((w8 + 255) / 256);
    cvt_f32_to_bf16<<<gin, blk, 0, stream>>>(query, xqb, in8);
    cvt_f32_to_bf16<<<gin, blk, 0, stream>>>(key_,  xkb, in8);
    cvt_f32_to_bf16<<<gin, blk, 0, stream>>>(value, xvb, in8);
    cvt_f32_to_bf16<<<gw,  blk, 0, stream>>>(wq, wqb, w8);
    cvt_f32_to_bf16<<<gw,  blk, 0, stream>>>(wk, wkb, w8);
    cvt_f32_to_bf16<<<gw,  blk, 0, stream>>>(wv, wvb, w8);
    cvt_f32_to_bf16<<<gw,  blk, 0, stream>>>(wo, wob, w8);

    const dim3 gproj(ROWS / 256, D_MODEL / 64, 1);   // 16 x 12
    gemm_xwT_bf16<<<gproj, blk, 0, stream>>>(xqb, D_MODEL, wqb, D_MODEL, bq, qb,  D_MODEL, D_MODEL, 1);
    gemm_xwT_bf16<<<gproj, blk, 0, stream>>>(xkb, D_MODEL, wkb, D_MODEL, bk, kb,  D_MODEL, D_MODEL, 1);
    gemm_xwT_bf16<<<gproj, blk, 0, stream>>>(xvb, D_MODEL, wvb, D_MODEL, bv, vtb, ROWS,    D_MODEL, 2);

    // fused scores+softmax+PV: one workgroup per (b, h, 16-query tile)
    const dim3 gat(SEQ / 16, 1, BATCH * N_HEADS);    // 128 x 1 x 24
    attn_fused<<<gat, blk, 0, stream>>>(qb, kb, vtb, mask, probs, aob);

    gemm_xwT_bf16<<<gproj, blk, 0, stream>>>(aob, D_MODEL, wob, D_MODEL, bo, out, D_MODEL, D_MODEL, 0);
}
